// CustomQwen3Attention_65128884076920
// MI455X (gfx1250) — compile-verified, matrix operand order checked by reading, not run on hardware
//
#include <hip/hip_runtime.h>

// ---------------------------------------------------------------------------
// Qwen3 attention block for MI455X (gfx1250, wave32, WMMA + TDM).
// B=1, S=2048, HID=4096, NH=32, NKV=8, HD=128. fp32 in/out, bf16 WMMA compute.
// ---------------------------------------------------------------------------

typedef __attribute__((ext_vector_type(16))) __bf16   v16bf;
typedef __attribute__((ext_vector_type(8)))  float    v8f;
typedef __attribute__((ext_vector_type(4)))  float    f32x4;
typedef __attribute__((ext_vector_type(8)))  unsigned v8u;
typedef __attribute__((ext_vector_type(4)))  unsigned u32x4;
typedef __attribute__((ext_vector_type(8)))  int      i32x8;
typedef __attribute__((ext_vector_type(4)))  int      i32x4;

#define S_LEN  2048
#define HIDDEN 4096
#define NH     32
#define NKV    8
#define HD     128
#define QKV_N  6144            // (NH + 2*NKV) * HD
#define RSCALE 0.08838834764831845f  // HD^-0.5
#define REPS   1e-6f

// ---------------------------------------------------------------------------
// WMMA helpers (A fragment layout per CDNA5 ISA 7.12.2; 32-bit LDS reads).
// ---------------------------------------------------------------------------
__device__ __forceinline__ v16bf frag_ld(const __bf16* base, int ld) {
  const int lane = threadIdx.x & 31;
  const unsigned* q = (const unsigned*)(base + (size_t)(lane & 15) * ld);
  const int kh2 = (lane >> 4) << 2;
  v8u f;
#pragma unroll
  for (int jj = 0; jj < 8; ++jj) {
    const int idx = (jj & 3) + ((jj >> 2) << 3) + kh2;
    f[jj] = q[idx];
  }
  return __builtin_bit_cast(v16bf, f);
}

__device__ __forceinline__ v8f wmma_bf16(v16bf a, v16bf b, v8f c) {
  return __builtin_amdgcn_wmma_f32_16x16x32_bf16(
      false, a, false, b, (short)0, c, false, false);
}

// ---------------------------------------------------------------------------
// Tensor Data Mover: 2D bf16 tile (global -> LDS) with LDS row padding.
// ---------------------------------------------------------------------------
__device__ __forceinline__ void tdm_load_2d(unsigned lds_off, const void* gp,
                                            unsigned tile_d0, unsigned tile_d1,
                                            unsigned stride0_elts,
                                            unsigned pad_interval_code,
                                            unsigned pad_amount_code) {
#if defined(__has_builtin) && __has_builtin(__builtin_amdgcn_tensor_load_to_lds)
  const unsigned long long ga = (unsigned long long)gp;
  u32x4 g0;
  g0[0] = 1u;                                    // count=1 (user descriptor)
  g0[1] = lds_off;                               // lds_addr
  g0[2] = (unsigned)(ga & 0xFFFFFFFFu);          // global_addr[31:0]
  g0[3] = (unsigned)((ga >> 32) & 0x01FFFFFFu) | (2u << 30);  // type=2
  i32x8 g1;
  g1[0] = (int)((1u << 16)                       // data_size = 2 bytes
              | (1u << 20)                       // pad_enable
              | (pad_interval_code << 22)
              | (pad_amount_code << 25));
  g1[1] = (int)((tile_d0 & 0xFFFFu) << 16);      // tensor_dim0[15:0]
  g1[2] = (int)((tile_d0 >> 16) | ((tile_d1 & 0xFFFFu) << 16));
  g1[3] = (int)((tile_d1 >> 16) | (tile_d0 << 16));   // dim1hi | tile_dim0
  g1[4] = (int)(tile_d1 & 0xFFFFu);              // tile_dim1 (tile_dim2 = 0)
  g1[5] = (int)stride0_elts;                     // tensor_dim0_stride[31:0]
  g1[6] = 0;
  g1[7] = 0;
  const i32x4 z4 = {0, 0, 0, 0};
#if __clang_major__ >= 23
  const i32x8 z8 = {0, 0, 0, 0, 0, 0, 0, 0};
  __builtin_amdgcn_tensor_load_to_lds(g0, g1, z4, z4, z8, 0);
#else
  __builtin_amdgcn_tensor_load_to_lds(g0, g1, z4, z4, 0);
#endif
#endif
}

// ---------------------------------------------------------------------------
// Kernel 1/4: C[M,N] = A[M,K] x B[K,N] (row-major f32), bf16 WMMA compute,
// f32 accumulate. Block tile 128x128, K-step 32, double-buffered LDS.
// Staging: 8 b128 loads issued in one batch (distinct register lifetimes),
// then convert/store, so the loads overlap the WMMA work on the other buffer.
// ---------------------------------------------------------------------------
__global__ __launch_bounds__(256) void gemm_rcr_bf16(
    const float* __restrict__ A, const float* __restrict__ B,
    float* __restrict__ C, int M, int N, int K) {
  __shared__ __bf16 sA[2][128][36];   // [buf][m][k]
  __shared__ __bf16 sB[2][128][36];   // [buf][n][k] (B staged transposed)
  const int tid  = threadIdx.x;
  const int lane = tid & 31;
  const int wave = tid >> 5;
  const int wm = (wave >> 1) * 32;
  const int wn = (wave & 1) * 64;
  const int bm = blockIdx.y * 128;
  const int bn = blockIdx.x * 128;

  const v8f vz = {0.f, 0.f, 0.f, 0.f, 0.f, 0.f, 0.f, 0.f};
  v8f acc[2][4];
#pragma unroll
  for (int mi = 0; mi < 2; ++mi)
#pragma unroll
    for (int ni = 0; ni < 4; ++ni) acc[mi][ni] = vz;

  auto stage = [&](int k0, int buf) {
    f32x4 ra[4], rb[4];
#pragma unroll
    for (int j = 0; j < 4; ++j) {                 // A: batch all loads
      const int idx = tid + j * 256;
      ra[j] = *(const f32x4*)&A[(size_t)(bm + (idx >> 3)) * K + k0 + (idx & 7) * 4];
    }
#pragma unroll
    for (int j = 0; j < 4; ++j) {                 // B: batch all loads
      const int idx = tid + j * 256;
      rb[j] = *(const f32x4*)&B[(size_t)(k0 + (idx >> 5)) * N + bn + (idx & 31) * 4];
    }
#pragma unroll
    for (int j = 0; j < 4; ++j) {                 // A: convert + b64 store
      const int idx = tid + j * 256;
      __bf16* d = &sA[buf][idx >> 3][(idx & 7) * 4];
      d[0] = (__bf16)ra[j].x; d[1] = (__bf16)ra[j].y;
      d[2] = (__bf16)ra[j].z; d[3] = (__bf16)ra[j].w;
    }
#pragma unroll
    for (int j = 0; j < 4; ++j) {                 // B: transposed stores
      const int idx = tid + j * 256;
      const int n4 = (idx & 31) * 4, c = idx >> 5;
      sB[buf][n4 + 0][c] = (__bf16)rb[j].x;
      sB[buf][n4 + 1][c] = (__bf16)rb[j].y;
      sB[buf][n4 + 2][c] = (__bf16)rb[j].z;
      sB[buf][n4 + 3][c] = (__bf16)rb[j].w;
    }
    if (k0 + 64 < K) {                            // prefetch 2 tiles ahead
      __builtin_prefetch(&A[(size_t)(bm + (tid >> 1)) * K + (k0 + 64)], 0, 1);
      __builtin_prefetch(&B[(size_t)(k0 + 64 + (tid >> 7)) * N + bn + (tid & 127)], 0, 1);
    }
  };

  stage(0, 0);
  __syncthreads();
  int buf = 0;
  for (int k0 = 0; k0 < K; k0 += 32) {
    if (k0 + 32 < K) stage(k0 + 32, buf ^ 1);     // overlap next tile

    v16bf bfr[4];
#pragma unroll
    for (int ni = 0; ni < 4; ++ni)
      bfr[ni] = frag_ld(&sB[buf][wn + ni * 16][0], 36);
#pragma unroll
    for (int mi = 0; mi < 2; ++mi) {
      const v16bf afr = frag_ld(&sA[buf][wm + mi * 16][0], 36);
#pragma unroll
      for (int ni = 0; ni < 4; ++ni)
        acc[mi][ni] = wmma_bf16(afr, bfr[ni], acc[mi][ni]);
    }
    __syncthreads();
    buf ^= 1;
  }

  const int rl = (lane >> 4) << 3;
  const int cl = lane & 15;
#pragma unroll
  for (int mi = 0; mi < 2; ++mi)
#pragma unroll
    for (int ni = 0; ni < 4; ++ni)
#pragma unroll
      for (int i = 0; i < 8; ++i)
        C[(size_t)(bm + wm + mi * 16 + i + rl) * N + (bn + wn + ni * 16 + cl)] =
            acc[mi][ni][i];
}

// ---------------------------------------------------------------------------
// Kernel 2/4: per-head RMSNorm (q,k) + NeoX RoPE + bf16 pack.
// Q/K head-major [H][S][HD]; V d-major [NKV][HD][S] (TDM tiles, no transpose).
// ---------------------------------------------------------------------------
__global__ __launch_bounds__(128) void norm_rope(
    const float* __restrict__ qkv, const int* __restrict__ pos,
    const float* __restrict__ cosc, const float* __restrict__ sinc,
    const float* __restrict__ qw, const float* __restrict__ kw,
    __bf16* __restrict__ Qo, __bf16* __restrict__ Ko, __bf16* __restrict__ Vo) {
  __shared__ float ssum[4];
  __shared__ float xbuf[128];
  const int s = blockIdx.x;
  const int t = threadIdx.x;
  const int p = pos[s];
  const int hx = t & 63;
  const float cs = cosc[(size_t)p * (HD / 2) + hx];
  const float sn = sinc[(size_t)p * (HD / 2) + hx];

  for (int h = 0; h < NH + 2 * NKV; ++h) {
    const float x = qkv[(size_t)s * QKV_N + h * HD + t];
    if (h < NH + NKV) {
      float ss = x * x;
#pragma unroll
      for (int off = 16; off > 0; off >>= 1) ss += __shfl_xor(ss, off, 32);
      if ((t & 31) == 0) ssum[t >> 5] = ss;
      __syncthreads();
      const float var = (ssum[0] + ssum[1] + ssum[2] + ssum[3]) * (1.f / HD);
      const float xn = x * rsqrtf(var + REPS) * (h < NH ? qw[t] : kw[t]);
      xbuf[t] = xn;
      __syncthreads();
      float out;
      if (t < 64) out = xn * cs - xbuf[t + 64] * sn;
      else        out = xn * cs + xbuf[t - 64] * sn;
      if (h < NH) Qo[((size_t)h * S_LEN + s) * HD + t] = (__bf16)out;
      else        Ko[((size_t)(h - NH) * S_LEN + s) * HD + t] = (__bf16)out;
      __syncthreads();
    } else {
      Vo[((size_t)(h - NH - NKV) * HD + t) * S_LEN + s] = (__bf16)x;
    }
  }
}

// ---------------------------------------------------------------------------
// Kernel 3/4: causal flash attention (GQA 4:1), bf16 WMMA.
// K/V tiles double-buffered via TDM: wave 0 issues tiles for iteration i+1,
// then s_wait_tensorcnt 2 (in-order completion => current pair resident while
// the next pair is still in flight). TDM pad feature produces the padded LDS
// leading dims (132 for Q/K, 36 for V) directly.
// ---------------------------------------------------------------------------
__global__ __launch_bounds__(256) void flash_attn(
    const __bf16* __restrict__ Q, const __bf16* __restrict__ K,
    const __bf16* __restrict__ V, float* __restrict__ Oa) {
  __shared__ __bf16 sQ[128][HD + 4];       // [qrow][d]        ld 132
  __shared__ __bf16 sK[2][32][HD + 4];     // [buf][key][d]    ld 132
  __shared__ __bf16 sV[2][HD][36];         // [buf][d][key]    ld 36
  __shared__ __bf16 sP[8][16][36];         // per-wave P tile  [qrow][key]
  const int tid  = threadIdx.x;
  const int lane = tid & 31;
  const int wave = tid >> 5;
  const int qb  = blockIdx.x;
  const int h   = blockIdx.y;
  const int kvh = h >> 2;                  // NH/NKV = 4
  const int q0  = qb * 128;
  const __bf16* Qh = Q + ((size_t)h * S_LEN + q0) * HD;
  const __bf16* Kh = K + (size_t)kvh * S_LEN * HD;
  const __bf16* Vt = V + (size_t)kvh * HD * S_LEN;   // d-major [HD][S]

  const unsigned ldsQ  = (unsigned)(unsigned long long)(void*)&sQ[0][0];
  const unsigned ldsK0 = (unsigned)(unsigned long long)(void*)&sK[0][0][0];
  const unsigned ldsK1 = (unsigned)(unsigned long long)(void*)&sK[1][0][0];
  const unsigned ldsV0 = (unsigned)(unsigned long long)(void*)&sV[0][0][0];
  const unsigned ldsV1 = (unsigned)(unsigned long long)(void*)&sV[1][0][0];

  const int nkb = q0 / 32 + 4;             // causal #key-blocks for this WG
  if (wave == 0) {                         // prologue: Q + first K/V pair
    tdm_load_2d(ldsQ, Qh, HD, 128, HD, 5, 1);
    tdm_load_2d(ldsK0, Kh, HD, 32, HD, 5, 1);
    tdm_load_2d(ldsV0, Vt, 32, HD, S_LEN, 3, 1);
  }

  const v8f vz = {0.f, 0.f, 0.f, 0.f, 0.f, 0.f, 0.f, 0.f};
  v8f oacc[8];
  float m8[8], l8[8];
#pragma unroll
  for (int d = 0; d < 8; ++d) oacc[d] = vz;
#pragma unroll
  for (int i = 0; i < 8; ++i) { m8[i] = -1e30f; l8[i] = 0.f; }

  const int qrw = q0 + wave * 16;
  const int rl  = (lane >> 4) << 3;
  const int cl  = lane & 15;

  for (int ib = 0; ib < nkb; ++ib) {
    const int kb  = ib * 32;
    const int buf = ib & 1;
    if (wave == 0) {
      if (ib + 1 < nkb) {                  // DMA next pair into other buffer
        tdm_load_2d(buf ? ldsK0 : ldsK1, Kh + (size_t)(kb + 32) * HD,
                    HD, 32, HD, 5, 1);
        tdm_load_2d(buf ? ldsV0 : ldsV1, Vt + kb + 32, 32, HD, S_LEN, 3, 1);
        __builtin_amdgcn_s_wait_tensorcnt(2);   // current pair resident
      } else {
        __builtin_amdgcn_s_wait_tensorcnt(0);
      }
    }
    __syncthreads();

    if (kb <= qrw + 15) {                  // wave-uniform causal skip
      v8f sc[2] = {vz, vz};
#pragma unroll
      for (int kc = 0; kc < 4; ++kc) {
        const v16bf qf = frag_ld(&sQ[wave * 16][kc * 32], HD + 4);
#pragma unroll
        for (int nt = 0; nt < 2; ++nt) {
          const v16bf kf = frag_ld(&sK[buf][nt * 16][kc * 32], HD + 4);
          sc[nt] = wmma_bf16(qf, kf, sc[nt]);
        }
      }
#pragma unroll
      for (int i = 0; i < 8; ++i) {        // online softmax
        const int qrow = qrw + i + rl;
        float s0 = sc[0][i] * RSCALE;
        float s1 = sc[1][i] * RSCALE;
        if (kb + cl      > qrow) s0 = -1e30f;
        if (kb + 16 + cl > qrow) s1 = -1e30f;
        float mx = fmaxf(s0, s1);
#pragma unroll
        for (int off = 1; off < 16; off <<= 1)
          mx = fmaxf(mx, __shfl_xor(mx, off, 32));
        const float mn = fmaxf(m8[i], mx);
        const float p0 = __expf(s0 - mn);
        const float p1 = __expf(s1 - mn);
        sP[wave][i + rl][cl]      = (__bf16)p0;
        sP[wave][i + rl][16 + cl] = (__bf16)p1;
        float rs = p0 + p1;
#pragma unroll
        for (int off = 1; off < 16; off <<= 1) rs += __shfl_xor(rs, off, 32);
        const float corr = __expf(m8[i] - mn);
        m8[i] = mn;
        l8[i] = l8[i] * corr + rs;
#pragma unroll
        for (int d = 0; d < 8; ++d) oacc[d][i] *= corr;
      }
      asm volatile("s_wait_dscnt 0x0" ::: "memory");  // same-wave LDS RAW
      const v16bf pf = frag_ld(&sP[wave][0][0], 36);
#pragma unroll
      for (int d = 0; d < 8; ++d) {
        const v16bf vf = frag_ld(&sV[buf][d * 16][0], 36);
        oacc[d] = wmma_bf16(pf, vf, oacc[d]);
      }
    }
    __syncthreads();
  }

#pragma unroll
  for (int i = 0; i < 8; ++i) {
    const float inv = l8[i] > 0.f ? 1.f / l8[i] : 0.f;
    const int row = qrw + i + rl;
#pragma unroll
    for (int d = 0; d < 8; ++d)
      Oa[(size_t)row * (NH * HD) + h * HD + d * 16 + cl] = oacc[d][i] * inv;
  }
}

// ---------------------------------------------------------------------------
// Host launcher. Workspace (bytes):
//   [0, 48MB)   qkv f32 (stage 1), aliased by attn f32 (stage 3, 32MB)
//   [48,64)MB   Q bf16 [NH][S][HD]
//   [64,68)MB   K bf16 [NKV][S][HD]
//   [68,72)MB   V bf16 [NKV][HD][S]
// ---------------------------------------------------------------------------
extern "C" void kernel_launch(void* const* d_in, const int* in_sizes, int n_in,
                              void* d_out, int out_size, void* d_ws,
                              size_t ws_size, hipStream_t stream) {
  const float* hidden = (const float*)d_in[0];
  const int*   posi   = (const int*)d_in[1];
  const float* cosc   = (const float*)d_in[2];
  const float* sinc   = (const float*)d_in[3];
  const float* w_qkv  = (const float*)d_in[4];
  const float* w_o    = (const float*)d_in[5];
  const float* qnw    = (const float*)d_in[6];
  const float* knw    = (const float*)d_in[7];

  char* ws = (char*)d_ws;
  float* qkv  = (float*)ws;
  float* attn = (float*)ws;
  const size_t qkv_bytes = (size_t)S_LEN * QKV_N * 4;
  __bf16* Qbf = (__bf16*)(ws + qkv_bytes);
  __bf16* Kbf = (__bf16*)(ws + qkv_bytes + (size_t)NH * S_LEN * HD * 2);
  __bf16* Vbf = (__bf16*)(ws + qkv_bytes + (size_t)NH * S_LEN * HD * 2 +
                          (size_t)NKV * S_LEN * HD * 2);
  float* out = (float*)d_out;

  dim3 g1(QKV_N / 128, S_LEN / 128);
  gemm_rcr_bf16<<<g1, 256, 0, stream>>>(hidden, w_qkv, qkv, S_LEN, QKV_N, HIDDEN);

  norm_rope<<<S_LEN, 128, 0, stream>>>(qkv, posi, cosc, sinc, qnw, knw,
                                       Qbf, Kbf, Vbf);

  dim3 g3(S_LEN / 128, NH);
  flash_attn<<<g3, 256, 0, stream>>>(Qbf, Kbf, Vbf, attn);

  dim3 g4(HIDDEN / 128, S_LEN / 128);
  gemm_rcr_bf16<<<g4, 256, 0, stream>>>(attn, w_o, out, S_LEN, HIDDEN, HIDDEN);
}
